// TorchWLKernel_22024592293889
// MI455X (gfx1250) — 8-bit matrix operand order repaired and checked by reading, compiled, not run on hardware
//
#include <hip/hip_runtime.h>

typedef unsigned long long u64;
typedef unsigned int       u32;
typedef unsigned char      u8;
typedef int v8i __attribute__((ext_vector_type(8)));

#define BG    64
#define NN    128
#define TT    (BG*NN)
#define SLOTS 16384
#define EMPTY 0xFFFFFFFFFFFFFFFFull
#define SALTC 0xD1B54A32D192ED03ull

__device__ __forceinline__ u64 mix64(u64 x) {
  u64 z = x + 0x9E3779B97F4A7C15ull;
  z = (z ^ (z >> 30)) * 0xBF58476D1CE4E5B9ull;
  z = (z ^ (z >> 27)) * 0x94D049BB133111EBull;
  return z ^ (z >> 31);
}

__global__ void k_init(const int* __restrict__ labels, u64* __restrict__ h,
                       int* __restrict__ Kacc) {
  int t = blockIdx.x * blockDim.x + threadIdx.x;
  if (t < TT)       h[t]    = mix64((u64)(unsigned)labels[t]);
  if (t < BG * BG)  Kacc[t] = 0;
}

__global__ void k_adjbits(const float* __restrict__ adj, u32* __restrict__ adjbits) {
  int t = blockIdx.x * blockDim.x + threadIdx.x;
  if (t >= TT) return;
  const float* row = adj + (size_t)t * NN;
  u32 w[4] = {0u, 0u, 0u, 0u};
  for (int j = 0; j < NN; ++j)
    if (row[j] > 0.5f) w[j >> 5] |= (1u << (j & 31));
  ((uint4*)adjbits)[t] = make_uint4(w[0], w[1], w[2], w[3]);
}

__global__ void k_wl(u64* __restrict__ h, const u32* __restrict__ adjbits) {
  __shared__ u64 m[NN];
  int b = blockIdx.x, i = threadIdx.x;
  u64 hold = h[b * NN + i];
  m[i] = mix64(hold ^ SALTC);
  __syncthreads();
  const u32* rb = adjbits + (size_t)(b * NN + i) * 4;
  u64 nsum = 0ull;
  #pragma unroll
  for (int w = 0; w < 4; ++w) {
    u32 bits = rb[w];
    while (bits) {
      int j = __builtin_ctz(bits);
      bits &= bits - 1u;
      nsum += m[w * 32 + j];
    }
  }
  h[b * NN + i] = mix64(mix64(hold) + nsum);
}

__global__ void k_clear(u64* __restrict__ keys, u32* __restrict__ counts32) {
  int t = blockIdx.x * blockDim.x + threadIdx.x;
  if (t < (BG * SLOTS) / 4) counts32[t] = 0u;
  if (t < SLOTS)            keys[t]     = EMPTY;
}

__global__ void k_hist(const u64* __restrict__ h, u64* __restrict__ keys,
                       u32* __restrict__ counts32) {
  int t = blockIdx.x * blockDim.x + threadIdx.x;
  if (t >= TT) return;
  u64 key = h[t];
  int g = t >> 7;
  u32 slot = ((u32)(key ^ (key >> 32))) & (SLOTS - 1);
  for (;;) {
    u64 prev = atomicCAS((unsigned long long*)&keys[slot], EMPTY, key);
    if (prev == EMPTY || prev == key) break;
    slot = (slot + 1) & (SLOTS - 1);
  }
  u32 boff = (u32)g * SLOTS + slot;
  atomicAdd(&counts32[boff >> 2], 1u << ((boff & 3u) * 8u));
}

__global__ void k_gemm(const u8* __restrict__ counts, int* __restrict__ Kacc) {
  int wave = blockIdx.x;
  int lane = threadIdx.x;
  int g0 = (wave >> 2) * 16;
  int g1 = (wave & 3) * 16;
  int half = lane >> 4;
  int m    = lane & 15;
  const u32* arow = (const u32*)(counts + (size_t)(g0 + m) * SLOTS);
  const u32* bcol = (const u32*)(counts + (size_t)(g1 + m) * SLOTS);
  v8i acc = {0, 0, 0, 0, 0, 0, 0, 0};
  for (int kb = 0; kb < SLOTS; kb += 64) {
    int kd = kb >> 2;
    v8i A, Bv;
    #pragma unroll
    for (int v = 0; v < 8; ++v)
      A[v] = (int)arow[kd + ((v >> 1) * 4) + (v & 1) + half * 2];
    #pragma unroll
    for (int v = 0; v < 8; ++v)
      Bv[v] = (int)bcol[kd + ((v >> 1) * 4) + (v & 1) + half * 2];
    acc = __builtin_amdgcn_wmma_i32_16x16x64_iu8(false, A, false, Bv, acc,
                                                 false, false);
  }
  #pragma unroll
  for (int r = 0; r < 8; ++r)
    Kacc[(g0 + r + half * 8) * BG + (g1 + m)] += acc[r];
}

__global__ void k_norm(const int* __restrict__ Kacc, float* __restrict__ out) {
  int t = blockIdx.x * blockDim.x + threadIdx.x;
  if (t >= BG * BG) return;
  int b1 = t >> 6, b2 = t & 63;
  float d1 = sqrtf((float)Kacc[b1 * BG + b1]);
  float d2 = sqrtf((float)Kacc[b2 * BG + b2]);
  out[t] = (float)Kacc[t] / (d1 * d2);
}

extern "C" void kernel_launch(void* const* d_in, const int* in_sizes, int n_in,
                              void* d_out, int out_size, void* d_ws, size_t ws_size,
                              hipStream_t stream) {
  const float* adj    = (const float*)d_in[0];
  const int*   labels = (const int*)d_in[1];
  float*       out    = (float*)d_out;

  char* ws = (char*)d_ws;
  u64* h        = (u64*)(ws + 0);
  u32* adjbits  = (u32*)(ws + 65536);
  u64* keys     = (u64*)(ws + 196608);
  u8*  counts   = (u8*) (ws + 327680);
  int* Kacc     = (int*)(ws + 1376256);

  k_init   <<<TT / 256, 256, 0, stream>>>(labels, h, Kacc);
  k_adjbits<<<TT / 256, 256, 0, stream>>>(adj, adjbits);

  for (int it = 0; it <= 5; ++it) {
    if (it > 0) k_wl<<<BG, NN, 0, stream>>>(h, adjbits);
    k_clear<<<(BG * SLOTS / 4) / 256, 256, 0, stream>>>(keys, (u32*)counts);
    k_hist <<<TT / 256, 256, 0, stream>>>(h, keys, (u32*)counts);
    k_gemm <<<16, 32, 0, stream>>>(counts, Kacc);
  }

  k_norm<<<(BG * BG) / 256, 256, 0, stream>>>(Kacc, out);
}
